// SynapticMoE_34497177322132
// MI455X (gfx1250) — matrix operand order repaired and checked by reading, not compiled, not run on hardware
//
#include <hip/hip_runtime.h>
#include <hip/hip_bf16.h>

// ---------------------------------------------------------------------------
// SynapticMoE for MI455X (gfx1250, wave32, WMMA f32_16x16x32_f16)
// Sparse top-2 dispatch; fused-weight (wf + mg*ws + hf) f16 GEMMs.
// Block tile: 64 tokens x 256 cols; wave tile: 32x64 (2 A-frags x 4 B-frags).
// ---------------------------------------------------------------------------

typedef __attribute__((ext_vector_type(16))) _Float16 v16h;
typedef __attribute__((ext_vector_type(8)))  _Float16 v8h;
typedef __attribute__((ext_vector_type(2)))  _Float16 h2;
typedef __attribute__((ext_vector_type(8)))  float    v8f;

#define NB   2
#define NT   2048
#define NC   768          // C
#define NH   3072         // H
#define NE   8            // experts
#define NTOK (NB * NT)    // 4096 tokens
#define NDR  24
#define MAXROWS 8704      // N*2 + 8*63 padding, rounded to 64 -> 136 tiles
#define BM   64           // token-tile rows per block
#define BN   256          // output cols per block
#define LDB  316          // LDS h2 stride per k-pair row; 8*LDB%64==32, mult of 4
#define SKW  20           // per-16-col-group skew (dwords), mult of 4 -> 16B stores

__device__ __forceinline__ float4 ld4(const float* p) { return *(const float4*)p; }

// ---------------------------------------------------------------------------
// Kernel 0: zero the output
// ---------------------------------------------------------------------------
__global__ void zero_kernel(float4* __restrict__ out, int n4) {
  int i = blockIdx.x * 256 + threadIdx.x;
  if (i < n4) out[i] = make_float4(0.f, 0.f, 0.f, 0.f);
}

// ---------------------------------------------------------------------------
// Kernel 1: router — one wave per token
// ---------------------------------------------------------------------------
__global__ __launch_bounds__(256)
void router_kernel(const float* __restrict__ x,
                   const float* __restrict__ rw,
                   const float* __restrict__ pw,
                   const float* __restrict__ remb,
                   const float* __restrict__ fatigue,
                   const float* __restrict__ energy,
                   int*   __restrict__ topk_idx,
                   float* __restrict__ topk_gate) {
  const int lane = threadIdx.x & 31;
  const int wid  = threadIdx.x >> 5;
  const int t    = blockIdx.x * 8 + wid;
  const float* xr = x + (size_t)t * NC;

  float xv[NC / 32];
  float rsum = 0.f;
#pragma unroll
  for (int j = 0; j < NC / 32; ++j) { xv[j] = xr[lane + 32 * j]; rsum += xv[j]; }
#pragma unroll
  for (int s = 16; s; s >>= 1) rsum += __shfl_xor(rsum, s, 32);
  const float mean = rsum * (1.0f / (float)NC);

  float logit[NE];
#pragma unroll
  for (int e = 0; e < NE; ++e) {
    const float* wr = rw + (size_t)e * NC;
    float acc = 0.f;
#pragma unroll
    for (int j = 0; j < NC / 32; ++j) acc += xv[j] * wr[lane + 32 * j];
#pragma unroll
    for (int s = 16; s; s >>= 1) acc += __shfl_xor(acc, s, 32);
    logit[e] = acc;
  }

  float pd[NDR];
#pragma unroll
  for (int d = 0; d < NDR; ++d) {
    const float* pr = pw + (size_t)d * NC;
    float acc = 0.f;
#pragma unroll
    for (int j = 0; j < NC / 32; ++j) acc += xv[j] * pr[lane + 32 * j];
#pragma unroll
    for (int s = 16; s; s >>= 1) acc += __shfl_xor(acc, s, 32);
    pd[d] = acc;
  }
  float pn = 0.f;
#pragma unroll
  for (int d = 0; d < NDR; ++d) pn += pd[d] * pd[d];
  const float inv_tok = 1.0f / fmaxf(sqrtf(pn), 1e-12f);

#pragma unroll
  for (int e = 0; e < NE; ++e) {
    float g2 = 0.f, dp = 0.f;
#pragma unroll
    for (int d = 0; d < NDR; ++d) {
      float ev = remb[e * NDR + d];
      g2 += ev * ev;
      dp += pd[d] * ev;
    }
    float gain = sqrtf(g2);
    float align = dp * inv_tok / fmaxf(gain, 1e-12f);
    logit[e] += 0.02f * mean + 0.02f * mean * gain + 0.02f * align
              + 0.1f * energy[e] - 0.1f * fatigue[e];
  }

  // top-2 (first-occurrence tie break like jax top_k)
  float g0 = -1e30f; int i0 = 0;
#pragma unroll
  for (int e = 0; e < NE; ++e) if (logit[e] > g0) { g0 = logit[e]; i0 = e; }
  float g1 = -1e30f; int i1 = 0;
#pragma unroll
  for (int e = 0; e < NE; ++e) if (e != i0 && logit[e] > g1) { g1 = logit[e]; i1 = e; }

  const float ex = expf(g1 - g0);
  const float gate0 = 1.0f / (1.0f + ex);
  const float gate1 = ex * gate0;

  if (lane == 0) {
    topk_idx[2 * t]      = i0;
    topk_idx[2 * t + 1]  = i1;
    topk_gate[2 * t]     = gate0;
    topk_gate[2 * t + 1] = gate1;
  }
}

// ---------------------------------------------------------------------------
// Kernel 2: build per-expert token lists (single block). Offsets 64-aligned.
// ---------------------------------------------------------------------------
__global__ __launch_bounds__(256)
void build_lists_kernel(const int* __restrict__ topk_idx,
                        const float* __restrict__ topk_gate,
                        int* __restrict__ offsets,
                        int* __restrict__ atok,
                        float* __restrict__ agate) {
  __shared__ int cnt[NE], cnt2[NE], offs[NE + 1];
  const int tid = threadIdx.x;
  if (tid < NE) { cnt[tid] = 0; cnt2[tid] = 0; }
  for (int i = tid; i < MAXROWS; i += 256) { atok[i] = 0; agate[i] = 0.f; }
  __syncthreads();
  for (int i = tid; i < NTOK * 2; i += 256) atomicAdd(&cnt[topk_idx[i]], 1);
  __syncthreads();
  if (tid == 0) {
    int off = 0;
    for (int e = 0; e < NE; ++e) { offs[e] = off; off += (cnt[e] + 63) & ~63; }
    offs[NE] = off;
    for (int e = 0; e <= NE; ++e) offsets[e] = offs[e];
  }
  __syncthreads();
  for (int i = tid; i < NTOK * 2; i += 256) {
    int e = topk_idx[i];
    int p = offs[e] + atomicAdd(&cnt2[e], 1);
    atok[p]  = i >> 1;
    agate[p] = topk_gate[i];
  }
}

// ---------------------------------------------------------------------------
// expert lookup helper (offsets are 64-aligned; BM=64 tiles never straddle)
// ---------------------------------------------------------------------------
__device__ __forceinline__ int expert_of(const int* __restrict__ offsets, int r0) {
  int e = 0;
#pragma unroll
  for (int i = 1; i < NE; ++i) if (r0 >= offsets[i]) e = i;
  return e;
}

// stage one fused 32(K)x256(N) weight tile into LDS as f16 k-pairs, skewed.
// thread role: kp = k-pair row (tid>>4), cg = 16-col group (tid&15).
// writes one 16B-aligned v8h per 4 columns -> ds_store_b128.
__device__ __forceinline__ void stage_wtile(h2* __restrict__ bs,
                                            const float* __restrict__ wf,
                                            const float* __restrict__ wsl,
                                            const float* __restrict__ hf,
                                            size_t ga, size_t gb, float mg,
                                            int kp, int cg) {
  h2* dst = &bs[kp * LDB + cg * SKW];
#pragma unroll
  for (int i = 0; i < 4; ++i) {
    float4 f0 = ld4(wf + ga + 4 * i), s0 = ld4(wsl + ga + 4 * i), q0 = ld4(hf + ga + 4 * i);
    float4 f1 = ld4(wf + gb + 4 * i), s1 = ld4(wsl + gb + 4 * i), q1 = ld4(hf + gb + 4 * i);
    v8h w;
    w[0] = (_Float16)(f0.x + mg * s0.x + q0.x);
    w[1] = (_Float16)(f1.x + mg * s1.x + q1.x);
    w[2] = (_Float16)(f0.y + mg * s0.y + q0.y);
    w[3] = (_Float16)(f1.y + mg * s1.y + q1.y);
    w[4] = (_Float16)(f0.z + mg * s0.z + q0.z);
    w[5] = (_Float16)(f1.z + mg * s1.z + q1.z);
    w[6] = (_Float16)(f0.w + mg * s0.w + q0.w);
    w[7] = (_Float16)(f1.w + mg * s1.w + q1.w);
    *(v8h*)(&dst[4 * i]) = w;   // 16B aligned: kp*LDB + cg*SKW + 4i, all mult of 4
  }
}

// load a B fragment (32x16 f16) from the skewed LDS tile
__device__ __forceinline__ v16h load_bfrag(const h2* __restrict__ bs,
                                           int hi, int grp, int lr) {
  v16h bf;
#pragma unroll
  for (int v = 0; v < 8; ++v) {
    h2 p = bs[(hi * 4 + v + (v >> 2) * 4) * LDB + grp * SKW + lr];
    bf[2 * v]     = p.x;
    bf[2 * v + 1] = p.y;
  }
  return bf;
}

// ---------------------------------------------------------------------------
// Kernel 3: FFN layer 1 — h = relu(x @ W1 + b1)^2, W1 = wf + mg*ws + hf
// grid (MAXROWS/BM, NH/BN), block 256 (8 waves: 2 M-halves x 4 N-quarters,
// each wave owns a 32x64 output tile = 8 WMMA accumulators)
// ---------------------------------------------------------------------------
__global__ __launch_bounds__(256)
void ffn1_kernel(const float* __restrict__ x,
                 const int* __restrict__ atok,
                 const int* __restrict__ offsets,
                 const float* __restrict__ wf, const float* __restrict__ wsl,
                 const float* __restrict__ hf, const float* __restrict__ mg1,
                 const float* __restrict__ b1,
                 _Float16* __restrict__ hbuf) {
  __shared__ h2 bs[16 * LDB];
  const int r0 = blockIdx.x * BM;
  if (r0 >= offsets[NE]) return;
  const int e  = expert_of(offsets, r0);
  const float mg = mg1[e];
  const int n0 = blockIdx.y * BN;

  const int tid  = threadIdx.x;
  const int lane = tid & 31;
  const int wid  = tid >> 5;
  const int mh   = wid & 1;         // which 32-row half
  const int wq   = wid >> 1;        // which 64-col quarter
  const int lr   = lane & 15;
  const int hi   = lane >> 4;

  const int tok0 = atok[r0 + mh * 32 + lr];
  const int tok1 = atok[r0 + mh * 32 + 16 + lr];
  const float* xrow0 = x + (size_t)tok0 * NC;
  const float* xrow1 = x + (size_t)tok1 * NC;
  const size_t wbase = (size_t)e * NC * NH;

  const int kp = tid >> 4;
  const int cg = tid & 15;

  v8f acc[2][4];
#pragma unroll
  for (int f = 0; f < 2; ++f)
#pragma unroll
    for (int s = 0; s < 4; ++s)
#pragma unroll
      for (int i = 0; i < 8; ++i) acc[f][s][i] = 0.0f;

  for (int k0 = 0; k0 < NC; k0 += 32) {
    __syncthreads();
    const size_t ga = wbase + (size_t)(k0 + 2 * kp) * NH + n0 + cg * 16;
    stage_wtile(bs, wf, wsl, hf, ga, ga + NH, mg, kp, cg);
    if (k0 + 32 < NC) {               // prefetch next K-tile (global_prefetch_b8)
      const size_t gn = ga + (size_t)32 * NH;
      __builtin_prefetch(wf + gn, 0, 3);
      __builtin_prefetch(wsl + gn, 0, 3);
      __builtin_prefetch(hf + gn, 0, 3);
      __builtin_prefetch(wf + gn + NH, 0, 3);
      __builtin_prefetch(wsl + gn + NH, 0, 3);
      __builtin_prefetch(hf + gn + NH, 0, 3);
    }
    __syncthreads();

    // two A fragments (16x32 f16 each), K-contiguous per lane:
    // lanes<16 hold K {0..7,16..23}, lanes>=16 {8..15,24..31}
    const int klo = k0 + hi * 8;
    v16h af[2];
#pragma unroll
    for (int f = 0; f < 2; ++f) {
      const float* xr = f ? xrow1 : xrow0;
      float4 a0 = ld4(xr + klo), a1 = ld4(xr + klo + 4);
      float4 c0 = ld4(xr + klo + 16), c1 = ld4(xr + klo + 20);
      af[f][0]  = (_Float16)a0.x; af[f][1]  = (_Float16)a0.y;
      af[f][2]  = (_Float16)a0.z; af[f][3]  = (_Float16)a0.w;
      af[f][4]  = (_Float16)a1.x; af[f][5]  = (_Float16)a1.y;
      af[f][6]  = (_Float16)a1.z; af[f][7]  = (_Float16)a1.w;
      af[f][8]  = (_Float16)c0.x; af[f][9]  = (_Float16)c0.y;
      af[f][10] = (_Float16)c0.z; af[f][11] = (_Float16)c0.w;
      af[f][12] = (_Float16)c1.x; af[f][13] = (_Float16)c1.y;
      af[f][14] = (_Float16)c1.z; af[f][15] = (_Float16)c1.w;
    }

#pragma unroll
    for (int sub = 0; sub < 4; ++sub) {
      v16h bf = load_bfrag(bs, hi, wq * 4 + sub, lr);
      acc[0][sub] = __builtin_amdgcn_wmma_f32_16x16x32_f16(
          false, af[0], false, bf, (short)0, acc[0][sub], false, false);
      acc[1][sub] = __builtin_amdgcn_wmma_f32_16x16x32_f16(
          false, af[1], false, bf, (short)0, acc[1][sub], false, false);
    }
  }

  // epilogue: relu^2, f16 store to hbuf
#pragma unroll
  for (int sub = 0; sub < 4; ++sub) {
    const int col = n0 + wq * 64 + sub * 16 + lr;
    const float bb = b1[e * NH + col];
#pragma unroll
    for (int f = 0; f < 2; ++f)
#pragma unroll
      for (int v = 0; v < 8; ++v) {
        const int row = r0 + mh * 32 + f * 16 + v + 8 * hi;
        float val = fmaxf(acc[f][sub][v] + bb, 0.f);
        hbuf[(size_t)row * NH + col] = (_Float16)(val * val);
      }
  }
}

// ---------------------------------------------------------------------------
// Kernel 4: FFN layer 2 — out += gate * (h @ W2 + b2), W2 = wf + mg*ws + hf
// grid (MAXROWS/BM, NC/BN), block 256
// ---------------------------------------------------------------------------
__global__ __launch_bounds__(256)
void ffn2_kernel(const _Float16* __restrict__ hbuf,
                 const int* __restrict__ atok,
                 const float* __restrict__ agate,
                 const int* __restrict__ offsets,
                 const float* __restrict__ wf, const float* __restrict__ wsl,
                 const float* __restrict__ hf, const float* __restrict__ mg2,
                 const float* __restrict__ b2,
                 float* __restrict__ out) {
  __shared__ h2 bs[16 * LDB];
  const int r0 = blockIdx.x * BM;
  if (r0 >= offsets[NE]) return;
  const int e  = expert_of(offsets, r0);
  const float mg = mg2[e];
  const int n0 = blockIdx.y * BN;

  const int tid  = threadIdx.x;
  const int lane = tid & 31;
  const int wid  = tid >> 5;
  const int mh   = wid & 1;
  const int wq   = wid >> 1;
  const int lr   = lane & 15;
  const int hi   = lane >> 4;

  const _Float16* hrow0 = hbuf + (size_t)(r0 + mh * 32 + lr) * NH;
  const _Float16* hrow1 = hbuf + (size_t)(r0 + mh * 32 + 16 + lr) * NH;
  const size_t wbase = (size_t)e * NH * NC;

  const int kp = tid >> 4;
  const int cg = tid & 15;

  v8f acc[2][4];
#pragma unroll
  for (int f = 0; f < 2; ++f)
#pragma unroll
    for (int s = 0; s < 4; ++s)
#pragma unroll
      for (int i = 0; i < 8; ++i) acc[f][s][i] = 0.0f;

  for (int k0 = 0; k0 < NH; k0 += 32) {
    __syncthreads();
    const size_t ga = wbase + (size_t)(k0 + 2 * kp) * NC + n0 + cg * 16;
    stage_wtile(bs, wf, wsl, hf, ga, ga + NC, mg, kp, cg);
    if (k0 + 32 < NH) {
      const size_t gn = ga + (size_t)32 * NC;
      __builtin_prefetch(wf + gn, 0, 3);
      __builtin_prefetch(wsl + gn, 0, 3);
      __builtin_prefetch(hf + gn, 0, 3);
      __builtin_prefetch(wf + gn + NC, 0, 3);
      __builtin_prefetch(wsl + gn + NC, 0, 3);
      __builtin_prefetch(hf + gn + NC, 0, 3);
    }
    __syncthreads();

    // A fragments straight from f16 hbuf: two 16B loads each (global_load_b128)
    const int klo = k0 + hi * 8;
    v8h alo0 = *(const v8h*)(hrow0 + klo);
    v8h ahi0 = *(const v8h*)(hrow0 + klo + 16);
    v8h alo1 = *(const v8h*)(hrow1 + klo);
    v8h ahi1 = *(const v8h*)(hrow1 + klo + 16);
    v16h af0 = __builtin_shufflevector(alo0, ahi0, 0, 1, 2, 3, 4, 5, 6, 7,
                                       8, 9, 10, 11, 12, 13, 14, 15);
    v16h af1 = __builtin_shufflevector(alo1, ahi1, 0, 1, 2, 3, 4, 5, 6, 7,
                                       8, 9, 10, 11, 12, 13, 14, 15);

#pragma unroll
    for (int sub = 0; sub < 4; ++sub) {
      v16h bf = load_bfrag(bs, hi, wq * 4 + sub, lr);
      acc[0][sub] = __builtin_amdgcn_wmma_f32_16x16x32_f16(
          false, af0, false, bf, (short)0, acc[0][sub], false, false);
      acc[1][sub] = __builtin_amdgcn_wmma_f32_16x16x32_f16(
          false, af1, false, bf, (short)0, acc[1][sub], false, false);
    }
  }

  // epilogue: gate-scaled atomic accumulate into output
  int tv[2][8]; float gv[2][8];
#pragma unroll
  for (int f = 0; f < 2; ++f)
#pragma unroll
    for (int v = 0; v < 8; ++v) {
      const int row = r0 + mh * 32 + f * 16 + v + 8 * hi;
      tv[f][v] = atok[row];
      gv[f][v] = agate[row];
    }
#pragma unroll
  for (int sub = 0; sub < 4; ++sub) {
    const int col = n0 + wq * 64 + sub * 16 + lr;
    const float bb = b2[e * NC + col];
#pragma unroll
    for (int f = 0; f < 2; ++f)
#pragma unroll
      for (int v = 0; v < 8; ++v) {
        const float val = (acc[f][sub][v] + bb) * gv[f][v];
        unsafeAtomicAdd(out + (size_t)tv[f][v] * NC + col, val);
      }
  }
}

// ---------------------------------------------------------------------------
// Host launcher
// ---------------------------------------------------------------------------
extern "C" void kernel_launch(void* const* d_in, const int* in_sizes, int n_in,
                              void* d_out, int out_size, void* d_ws, size_t ws_size,
                              hipStream_t stream) {
  (void)in_sizes; (void)n_in; (void)out_size; (void)ws_size;
  const float* x        = (const float*)d_in[0];
  const float* router_w = (const float*)d_in[1];
  const float* probe_w  = (const float*)d_in[2];
  const float* r_emb    = (const float*)d_in[3];
  const float* fatigue  = (const float*)d_in[4];
  const float* energy   = (const float*)d_in[5];
  const float* w_fast1  = (const float*)d_in[6];
  const float* w_slow1  = (const float*)d_in[7];
  const float* h_fast1  = (const float*)d_in[8];
  const float* m_gate1  = (const float*)d_in[9];
  const float* b1       = (const float*)d_in[10];
  const float* w_fast2  = (const float*)d_in[11];
  const float* w_slow2  = (const float*)d_in[12];
  const float* h_fast2  = (const float*)d_in[13];
  const float* m_gate2  = (const float*)d_in[14];
  const float* b2       = (const float*)d_in[15];
  float* out = (float*)d_out;

  // workspace layout (bytes); total ~53.6 MB
  char* ws = (char*)d_ws;
  int*      offsets   = (int*)(ws + 0);                 // 9 ints (pad 256)
  int*      topk_idx  = (int*)(ws + 256);               // 2N ints   = 32 KB
  float*    topk_gate = (float*)(ws + 256 + 32768);     // 2N floats = 32 KB
  int*      atok      = (int*)(ws + 65792);             // MAXROWS ints  (34816 B)
  float*    agate     = (float*)(ws + 100608);          // MAXROWS floats(34816 B)
  _Float16* hbuf      = (_Float16*)(ws + 135424);       // MAXROWS*NH f16 = 53.5 MB

  const int n4 = NTOK * NC / 4;
  zero_kernel<<<(n4 + 255) / 256, 256, 0, stream>>>((float4*)out, n4);
  router_kernel<<<NTOK / 8, 256, 0, stream>>>(x, router_w, probe_w, r_emb,
                                              fatigue, energy, topk_idx, topk_gate);
  build_lists_kernel<<<1, 256, 0, stream>>>(topk_idx, topk_gate, offsets, atok, agate);
  ffn1_kernel<<<dim3(MAXROWS / BM, NH / BN), 256, 0, stream>>>(
      x, atok, offsets, w_fast1, w_slow1, h_fast1, m_gate1, b1, hbuf);
  ffn2_kernel<<<dim3(MAXROWS / BM, NC / BN), 256, 0, stream>>>(
      hbuf, atok, agate, offsets, w_fast2, w_slow2, h_fast2, m_gate2, b2, out);
}
